// NetworkSubjectLayers_19310172963545
// MI455X (gfx1250) — matrix operand order repaired and checked by reading, not compiled, not run on hardware
//
#include <hip/hip_runtime.h>

typedef __attribute__((ext_vector_type(16))) __bf16       v16bf;
typedef __attribute__((ext_vector_type(8)))  float        v8f;
typedef __attribute__((ext_vector_type(4)))  float        v4f;
typedef __attribute__((ext_vector_type(4)))  unsigned int v4u;

struct Frag32 { v4u a, b; };   // 32 bytes == v16bf

#define KC    32     // K-chunk (bf16 WMMA depth)
#define MT    64     // block M tile (4 waves x 2 subtiles? -> 2 M-waves x 32)
#define NT    128    // block N tile (4 N-waves x 32)
#define ASTR2 20     // row stride in u32: 32 bf16 + 8 pad = 80B (16B aligned)

__device__ __forceinline__ unsigned short f32_to_bf16_rne(float f) {
  unsigned int u = __builtin_bit_cast(unsigned int, f);
  u += 0x7FFFu + ((u >> 16) & 1u);
  return (unsigned short)(u >> 16);
}
__device__ __forceinline__ float bf16_as_f32(unsigned short h) {
  unsigned int u = ((unsigned int)h) << 16;
  return __builtin_bit_cast(float, u);
}
__device__ __forceinline__ void split_bf16(float f, unsigned short& hi, unsigned short& lo) {
  unsigned short h = f32_to_bf16_rne(f);
  hi = h;
  lo = f32_to_bf16_rne(f - bf16_as_f32(h));
}
__device__ __forceinline__ unsigned int pack2(unsigned short lo16, unsigned short hi16) {
  return (unsigned int)lo16 | ((unsigned int)hi16 << 16);
}

// Split two floats into packed-bf16 (hi pair, lo pair). Uses v_cvt_pk_bf16_f32
// when the toolchain exposes it; manual RNE fallback otherwise.
__device__ __forceinline__ void split2_pack(float f0, float f1,
                                            unsigned int& hiPk, unsigned int& loPk) {
#if defined(__HIP_DEVICE_COMPILE__) && __has_builtin(__builtin_amdgcn_cvt_pk_bf16_f32)
  auto h = __builtin_amdgcn_cvt_pk_bf16_f32(f0, f1);
  unsigned int hp = __builtin_bit_cast(unsigned int, h);
  float h0 = __builtin_bit_cast(float, hp << 16);
  float h1 = __builtin_bit_cast(float, hp & 0xffff0000u);
  auto l = __builtin_amdgcn_cvt_pk_bf16_f32(f0 - h0, f1 - h1);
  hiPk = hp;
  loPk = __builtin_bit_cast(unsigned int, l);
#else
  unsigned short h0, l0, h1, l1;
  split_bf16(f0, h0, l0);
  split_bf16(f1, h1, l1);
  hiPk = pack2(h0, h1);
  loPk = pack2(l0, l1);
#endif
}

// ---- pass 0: zero per-subject counters ----
__global__ void zero_counts(int* counts, int n) {
  int i = blockIdx.x * blockDim.x + threadIdx.x;
  if (i < n) counts[i] = 0;
}

// ---- pass 1: counting-sort rows by subject (order within bucket irrelevant) ----
__global__ void bucket_rows(const int* __restrict__ sid, int* counts,
                            int* sorted, int B, int capB) {
  int i = blockIdx.x * blockDim.x + threadIdx.x;
  if (i < B) {
    int s = sid[i];
    int p = atomicAdd(&counts[s], 1);
    sorted[s * capB + p] = i;
  }
}

// ---- pass 2: per-subject gathered GEMM, bf16 hi/lo split, f32 accumulate ----
__global__ __launch_bounds__(256)
void subj_gemm(const float* __restrict__ x,
               const float* __restrict__ W,
               const float* __restrict__ bias,
               const int* __restrict__ sorted,
               const int* __restrict__ counts,
               float* __restrict__ out,
               int IN_CH, int OUT_DIM, int capB)
{
  // double-buffered bf16 tiles, packed u32 pairs (k even|odd), row stride 80B
  __shared__ __align__(16) unsigned int aHi[2][MT][ASTR2];
  __shared__ __align__(16) unsigned int aLo[2][MT][ASTR2];
  __shared__ __align__(16) unsigned int bHi[2][NT][ASTR2];
  __shared__ __align__(16) unsigned int bLo[2][NT][ASTR2];
  __shared__ int rowIdx[MT];

  const int s       = blockIdx.z;
  const int cnt     = counts[s];
  const int rowBase = blockIdx.y * MT;
  if (rowBase >= cnt) return;                 // uniform early-exit
  const int colBase = blockIdx.x * NT;
  const int tid     = threadIdx.x;

  const int* sortedS = sorted + (size_t)s * capB;
  if (tid < MT) {
    int p = rowBase + tid;
    rowIdx[tid] = (p < cnt) ? sortedS[p] : -1;
  }
  __syncthreads();

  const int lane = tid & 31;
  const int wv   = tid >> 5;
  const int wm   = wv & 1;        // wave M index (0..1), 32 rows each
  const int wn   = wv >> 1;       // wave N index (0..3), 32 cols each
  const int l15  = lane & 15;
  const int kbA2 = (lane >> 4) * 8;    // A frag u32 offset: K {kb..kb+15} contiguous
  const int kbB2 = (lane >> 4) * 8;    // B frag u32 offset: K {kb..kb+15} contiguous

  v8f acc[4];                          // [mt*2 + nt], each 16x16
  {
    v8f z = {};
    #pragma unroll
    for (int i = 0; i < 4; ++i) acc[i] = z;
  }

  // loader decomposition
  const int arg = tid >> 3;         // 0..31  (A row group)
  const int akk = (tid & 7) * 4;    // A k offset (float4, 4 consecutive k)
  int kpv[2], ogv[2];               // B loader: 16 k-pairs x 32 col-groups = 512 tasks
  bool bfull[2];
  #pragma unroll
  for (int t = 0; t < 2; ++t) {
    int t2 = tid + 256 * t;
    kpv[t] = t2 >> 5;               // 0..15 (k pair)
    ogv[t] = (t2 & 31) * 4;         // 0..124 (col group)
    bfull[t] = (colBase + ogv[t] + 3 < OUT_DIM);
  }

  const float* Ws = W + (size_t)s * IN_CH * OUT_DIM;

  // pipeline registers (current chunk's raw f32 data)
  v4f aR[2];
  v4f b0R[2], b1R[2];

  auto loadA = [&](int kc) {
    #pragma unroll
    for (int rr = 0; rr < 2; ++rr) {
      int row = arg + 32 * rr;
      int g   = rowIdx[row];
      v4f v   = {};
      if (g >= 0) v = *(const v4f*)(x + (size_t)g * IN_CH + kc + akk);
      aR[rr] = v;
    }
  };
  auto loadB = [&](int kc) {
    #pragma unroll
    for (int t = 0; t < 2; ++t) {
      const float* wrow0 = Ws + (size_t)(kc + 2 * kpv[t]) * OUT_DIM + colBase + ogv[t];
      const float* wrow1 = wrow0 + OUT_DIM;
      v4f v0 = {}, v1 = {};
      if (bfull[t]) {
        v0 = *(const v4f*)wrow0;
        v1 = *(const v4f*)wrow1;
      } else {
        #pragma unroll
        for (int j = 0; j < 4; ++j)
          if (colBase + ogv[t] + j < OUT_DIM) { v0[j] = wrow0[j]; v1[j] = wrow1[j]; }
      }
      b0R[t] = v0;
      b1R[t] = v1;
    }
  };

  // prologue: issue loads for chunk 0
  loadA(0);
  loadB(0);

  int pb = 0;
  for (int kc = 0; kc < IN_CH; kc += KC, pb ^= 1) {
    unsigned int (*aH)[ASTR2] = aHi[pb];
    unsigned int (*aL)[ASTR2] = aLo[pb];
    unsigned int (*bH)[ASTR2] = bHi[pb];
    unsigned int (*bL)[ASTR2] = bLo[pb];

    // ---- convert current regs -> bf16 hi/lo, packed b32 LDS stores ----
    #pragma unroll
    for (int rr = 0; rr < 2; ++rr) {
      int row = arg + 32 * rr;
      unsigned int hp0, lp0, hp1, lp1;
      split2_pack(aR[rr][0], aR[rr][1], hp0, lp0);
      split2_pack(aR[rr][2], aR[rr][3], hp1, lp1);
      aH[row][akk / 2]     = hp0;
      aH[row][akk / 2 + 1] = hp1;
      aL[row][akk / 2]     = lp0;
      aL[row][akk / 2 + 1] = lp1;
    }
    #pragma unroll
    for (int t = 0; t < 2; ++t) {
      #pragma unroll
      for (int j = 0; j < 4; ++j) {
        unsigned int hp, lp;
        split2_pack(b0R[t][j], b1R[t][j], hp, lp);   // {k, k+1} packed ascending
        bH[ogv[t] + j][kpv[t]] = hp;
        bL[ogv[t] + j][kpv[t]] = lp;
      }
    }
    __syncthreads();   // single barrier per chunk (double-buffered LDS)

    // ---- issue next chunk's global loads before compute (latency hiding) ----
    if (kc + KC < IN_CH) {
      loadA(kc + KC);
      loadB(kc + KC);
    }

    // ---- compute: 2x2 subtiles, 3 WMMA each (hi*hi + hi*lo + lo*hi) ----
    v16bf ahi[2], alo[2];
    #pragma unroll
    for (int mt = 0; mt < 2; ++mt) {
      int am = wm * 32 + mt * 16 + l15;
      Frag32 fh { *(const v4u*)&aH[am][kbA2], *(const v4u*)&aH[am][kbA2 + 4] };
      Frag32 fl { *(const v4u*)&aL[am][kbA2], *(const v4u*)&aL[am][kbA2 + 4] };
      ahi[mt] = __builtin_bit_cast(v16bf, fh);
      alo[mt] = __builtin_bit_cast(v16bf, fl);
    }
    #pragma unroll
    for (int nt = 0; nt < 2; ++nt) {
      int o = wn * 32 + nt * 16 + l15;
      Frag32 fh { *(const v4u*)&bH[o][kbB2], *(const v4u*)&bH[o][kbB2 + 4] };
      Frag32 fl { *(const v4u*)&bL[o][kbB2], *(const v4u*)&bL[o][kbB2 + 4] };
      v16bf bhi = __builtin_bit_cast(v16bf, fh);
      v16bf blo = __builtin_bit_cast(v16bf, fl);
      #pragma unroll
      for (int mt = 0; mt < 2; ++mt) {
        int ai = mt * 2 + nt;
        acc[ai] = __builtin_amdgcn_wmma_f32_16x16x32_bf16(false, ahi[mt], false, bhi,
                                                          (short)0, acc[ai], false, false);
        acc[ai] = __builtin_amdgcn_wmma_f32_16x16x32_bf16(false, ahi[mt], false, blo,
                                                          (short)0, acc[ai], false, false);
        acc[ai] = __builtin_amdgcn_wmma_f32_16x16x32_bf16(false, alo[mt], false, bhi,
                                                          (short)0, acc[ai], false, false);
      }
    }
  }

  // ---- scatter back to original row positions, add per-subject bias ----
  const int mo = (lane >> 4) * 8;
  #pragma unroll
  for (int nt = 0; nt < 2; ++nt) {
    int col = colBase + wn * 32 + nt * 16 + l15;
    if (col >= OUT_DIM) continue;
    float bv = bias[(size_t)s * OUT_DIM + col];
    #pragma unroll
    for (int mt = 0; mt < 2; ++mt) {
      #pragma unroll
      for (int r = 0; r < 8; ++r) {
        int row = wm * 32 + mt * 16 + mo + r;
        int g   = rowIdx[row];
        if (g >= 0) out[(size_t)g * OUT_DIM + col] = acc[mt * 2 + nt][r] + bv;
      }
    }
  }
}

extern "C" void kernel_launch(void* const* d_in, const int* in_sizes, int n_in,
                              void* d_out, int out_size, void* d_ws, size_t ws_size,
                              hipStream_t stream) {
  (void)n_in; (void)ws_size;
  const float* x    = (const float*)d_in[0];
  const int*   sid  = (const int*)d_in[1];
  const float* W    = (const float*)d_in[2];
  const float* bias = (const float*)d_in[3];
  float*       out  = (float*)d_out;

  const int B       = in_sizes[1];            // 8192
  const int IN_CH   = in_sizes[0] / B;        // 2048
  const int OUT_DIM = out_size / B;           // 1000
  const int NSUB    = in_sizes[3] / OUT_DIM;  // 8

  int* counts = (int*)d_ws;          // 16 ints
  int* sorted = counts + 16;         // NSUB * B ints (capacity B per subject)

  zero_counts<<<1, 32, 0, stream>>>(counts, 16);
  bucket_rows<<<(B + 255) / 256, 256, 0, stream>>>(sid, counts, sorted, B, B);

  dim3 grid((OUT_DIM + NT - 1) / NT, (B + MT - 1) / MT, NSUB);
  subj_gemm<<<grid, 256, 0, stream>>>(x, W, bias, sorted, counts, out,
                                      IN_CH, OUT_DIM, B);
}
